// FrontierReasoningExpertHead_33784212750947
// MI455X (gfx1250) — matrix operand order repaired and checked by reading, not compiled, not run on hardware
//
#include <hip/hip_runtime.h>
#include <cstdint>
#include <cstddef>

typedef __attribute__((ext_vector_type(16))) __bf16 v16bf;
typedef __attribute__((ext_vector_type(8)))  float  v8f;
typedef __attribute__((ext_vector_type(4)))  unsigned int u32x4;
typedef __attribute__((ext_vector_type(8)))  int  i32x8;
typedef __attribute__((ext_vector_type(4)))  int  i32x4;
typedef unsigned int u32;

#if defined(__has_builtin)
#if __has_builtin(__builtin_amdgcn_tensor_load_to_lds)
#define TDM_OK 1
#else
#define TDM_OK 0
#endif
#else
#define TDM_OK 0
#endif

// ---------------------------------------------------------------------------
// Activation library (ids match reference _ACTS order; 8 = identity)
// ---------------------------------------------------------------------------
__device__ __forceinline__ float apply_act(float x, int a)
{
    switch (a) {
    case 0: return x / (1.f + __expf(-x));                              // silu
    case 1: return 0.5f * x * (1.f + erff(x * 0.70710678118654752f));   // gelu (exact)
    case 2: { float sp = (x > 20.f) ? x : log1pf(__expf(x));            // mish
              return x * tanhf(sp); }
    case 3: return fmaxf(x, 0.f);                                       // relu
    case 4: return (x > 0.f) ? 1.0507009873554805f * x                  // selu
                 : 1.0507009873554805f * 1.6732632423543772f * (__expf(x) - 1.f);
    case 5: return tanhf(x);                                            // tanh
    case 6: return (x > 20.f) ? x : log1pf(__expf(x));                  // softplus
    case 7: return (x > 0.f) ? x : (__expf(x) - 1.f);                   // elu
    default: return x;                                                  // identity
    }
}

// ---------------------------------------------------------------------------
// WMMA GEMM v2:  C[M,N] = act(A[M,K] * Bt[N,K]^T + bias[N])
//   A,Bt bf16 row-major; C fp32; optional simultaneous bf16 output Cb.
//   Block 256 threads = 8 waves. Tile 128(M) x 64(N), K chunk 32.
//   Double-buffered LDS, one barrier per chunk, 4 WMMA/wave/chunk
//   sharing one A fragment (reuse_a hint on 2..4).
//   Requires M%128==0, N%64==0, K%32==0.
// ---------------------------------------------------------------------------
#define GTM 128
#define GTN 64
#define GTK 32

union FragU { uint4 q[2]; v16bf v; };

__global__ __launch_bounds__(256)
void k_wmma_gemm_bf16(const __bf16* __restrict__ A,
                      const __bf16* __restrict__ Bt,
                      float* __restrict__ C,
                      __bf16* __restrict__ Cb,
                      const float* __restrict__ bias,
                      int act, int M, int N, int K)
{
    (void)M;
    __shared__ __bf16 sA[2][GTM][GTK];
    __shared__ __bf16 sB[2][GTN][GTK];

    const int tid  = threadIdx.x;
    const int wave = tid >> 5;
    const int lane = tid & 31;
    const int m0   = blockIdx.y * GTM;
    const int n0   = blockIdx.x * GTN;
    const int nch  = K / GTK;

    v8f acc[4];
    #pragma unroll
    for (int s = 0; s < 4; ++s) acc[s] = (v8f){};

    const int ar0 = tid >> 2;            // A stage rows (0..63 / 64..127)
    const int ac0 = (tid & 3) * 8;

    // ---- prologue: stage chunk 0 into buffer 0
    *(uint4*)&sA[0][ar0][ac0]      = *(const uint4*)&A[(size_t)(m0 + ar0) * K + ac0];
    *(uint4*)&sA[0][ar0 + 64][ac0] = *(const uint4*)&A[(size_t)(m0 + ar0 + 64) * K + ac0];
    *(uint4*)&sB[0][ar0][ac0]      = *(const uint4*)&Bt[(size_t)(n0 + ar0) * K + ac0];
    __syncthreads();

    const int mrow = wave * 16 + (lane & 15);
    const int khi  = (lane >> 4) * 8;    // 16-bit A 16x32 VGPR layout
    const int ncol = lane & 15;
    const int kb   = (lane >> 4) * 16;   // unused; B loads use khi, same k order as A

    for (int i = 0; i < nch; ++i) {
        const int p = i & 1;

        // (1) fragment loads from buffer p (issued before staging stores)
        FragU af;
        af.q[0] = *(const uint4*)&sA[p][mrow][khi];
        af.q[1] = *(const uint4*)&sA[p][mrow][16 + khi];
        FragU bf[4];
        #pragma unroll
        for (int s = 0; s < 4; ++s) {
            bf[s].q[0] = *(const uint4*)&sB[p][s * 16 + ncol][khi];
            bf[s].q[1] = *(const uint4*)&sB[p][s * 16 + ncol][16 + khi];
        }

        // (2)+(3) stage chunk i+1 into buffer p^1
        if (i + 1 < nch) {
            const int kn = (i + 1) * GTK;
            uint4 ra0 = *(const uint4*)&A[(size_t)(m0 + ar0) * K + kn + ac0];
            uint4 ra1 = *(const uint4*)&A[(size_t)(m0 + ar0 + 64) * K + kn + ac0];
            uint4 rb  = *(const uint4*)&Bt[(size_t)(n0 + ar0) * K + kn + ac0];
            if (i + 2 < nch)
                __builtin_prefetch(&A[(size_t)(m0 + ar0) * K + kn + GTK + ac0], 0, 3);
            *(uint4*)&sA[p ^ 1][ar0][ac0]      = ra0;
            *(uint4*)&sA[p ^ 1][ar0 + 64][ac0] = ra1;
            *(uint4*)&sB[p ^ 1][ar0][ac0]      = rb;
        }

        // (4) matrix ops: one A fragment reused across 4 B fragments
        acc[0] = __builtin_amdgcn_wmma_f32_16x16x32_bf16(false, af.v, false, bf[0].v,
                                                         (short)0, acc[0], false, false);
        acc[1] = __builtin_amdgcn_wmma_f32_16x16x32_bf16(false, af.v, false, bf[1].v,
                                                         (short)0, acc[1], true, false);
        acc[2] = __builtin_amdgcn_wmma_f32_16x16x32_bf16(false, af.v, false, bf[2].v,
                                                         (short)0, acc[2], true, false);
        acc[3] = __builtin_amdgcn_wmma_f32_16x16x32_bf16(false, af.v, false, bf[3].v,
                                                         (short)0, acc[3], true, false);
        __syncthreads();
    }

    // ---- fused epilogue: bias + act, fp32 store, optional bf16 store
    const int crow = m0 + wave * 16 + (lane >> 4) * 8;
    const int cc0  = n0 + (lane & 15);
    #pragma unroll
    for (int s = 0; s < 4; ++s) {
        const int cc = cc0 + s * 16;
        const float bb = bias ? bias[cc] : 0.f;
        #pragma unroll
        for (int r = 0; r < 8; ++r) {
            float v = apply_act(acc[s][r] + bb, act);
            C[(size_t)(crow + r) * N + cc] = v;
            if (Cb) Cb[(size_t)(crow + r) * N + cc] = (__bf16)v;
        }
    }
}

// ---------------------------------------------------------------------------
// Tensor Data Mover descriptor issue: 2-D tile (tileRows x 32 bf16) from a
// row-major [rows x tensK] bf16 tensor into LDS. D# layout per ISA 8.3/8.4.
// ---------------------------------------------------------------------------
__device__ __forceinline__ void tdm_issue(void* lds, const void* g,
                                          unsigned tensK, unsigned rows,
                                          unsigned tileRows)
{
#if TDM_OK
    unsigned lptr = (unsigned)(unsigned long long)lds;
    unsigned long long ga = (unsigned long long)g;
    u32x4 g0;
    g0[0] = 1u;                                              // count=1, user D#
    g0[1] = lptr;                                            // lds_addr
    g0[2] = (unsigned)(ga & 0xFFFFFFFFu);                    // global_addr lo
    g0[3] = (unsigned)((ga >> 32) & 0x01FFFFFFu) | (2u << 30); // addr hi | type=2
    i32x8 g1;
    g1[0] = (int)(1u << 16);                                 // data_size=1 (2B)
    g1[1] = (int)((tensK & 0xFFFFu) << 16);                  // tensor_dim0 lo
    g1[2] = (int)(((tensK >> 16) & 0xFFFFu) | ((rows & 0xFFFFu) << 16));
    g1[3] = (int)(((rows >> 16) & 0xFFFFu) | ((unsigned)GTK << 16)); // tile_dim0=32
    g1[4] = (int)(tileRows & 0xFFFFu);                       // tile_dim1 (dim2=0)
    g1[5] = (int)tensK;                                      // dim0_stride lo
    g1[6] = 0;                                               // stride hi, dim1_stride
    g1[7] = 0;
    i32x4 z4 = {0, 0, 0, 0};
#if __clang_major__ >= 23
    i32x8 z8 = {0, 0, 0, 0, 0, 0, 0, 0};
    __builtin_amdgcn_tensor_load_to_lds(g0, g1, z4, z4, z8, 0);
#else
    __builtin_amdgcn_tensor_load_to_lds(g0, g1, z4, z4, 0);
#endif
#else
    (void)lds; (void)g; (void)tensK; (void)rows; (void)tileRows;
#endif
}

// ---------------------------------------------------------------------------
// TDM-fed WMMA GEMM (used for the dominant expert up-projections).
// Same tiling as above; LDS filled by tensor_load_to_lds, double-buffered,
// TENSORcnt-throttled. Falls back to cooperative loads if TDM is absent.
// ---------------------------------------------------------------------------
__global__ __launch_bounds__(256)
void k_wmma_gemm_tdm(const __bf16* __restrict__ A,
                     const __bf16* __restrict__ Bt,
                     float* __restrict__ C,
                     __bf16* __restrict__ Cb,
                     const float* __restrict__ bias,
                     int act, int M, int N, int K)
{
    (void)M;
    __shared__ __bf16 sA[2][GTM][GTK];
    __shared__ __bf16 sB[2][GTN][GTK];

    const int tid  = threadIdx.x;
    const int wave = tid >> 5;
    const int lane = tid & 31;
    const int m0   = blockIdx.y * GTM;
    const int n0   = blockIdx.x * GTN;
    const int nch  = K / GTK;

    v8f acc[4];
    #pragma unroll
    for (int s = 0; s < 4; ++s) acc[s] = (v8f){};

    const int mrow = wave * 16 + (lane & 15);
    const int khi  = (lane >> 4) * 8;
    const int ncol = lane & 15;
    const int kb   = (lane >> 4) * 16;

#if TDM_OK
    // prologue: issue chunk 0 into buffer 0 (wave 0 only; EXEC ignored by TDM)
    if (wave == 0) {
        tdm_issue(&sA[0][0][0], A + (size_t)m0 * K, (unsigned)K, (unsigned)GTM, GTM);
        tdm_issue(&sB[0][0][0], Bt + (size_t)n0 * K, (unsigned)K, (unsigned)GTN, GTN);
    }
    for (int i = 0; i < nch; ++i) {
        const int p = i & 1;
        if (wave == 0) {
            if (i + 1 < nch) {
                const int kn = (i + 1) * GTK;
                tdm_issue(&sA[p ^ 1][0][0], A + (size_t)m0 * K + kn, (unsigned)K,
                          (unsigned)GTM, GTM);
                tdm_issue(&sB[p ^ 1][0][0], Bt + (size_t)n0 * K + kn, (unsigned)K,
                          (unsigned)GTN, GTN);
                __builtin_amdgcn_s_wait_tensorcnt(2);   // chunk i complete
            } else {
                __builtin_amdgcn_s_wait_tensorcnt(0);
            }
        }
        __syncthreads();                                // chunk i visible to all

        FragU af;
        af.q[0] = *(const uint4*)&sA[p][mrow][khi];
        af.q[1] = *(const uint4*)&sA[p][mrow][16 + khi];
        FragU bf[4];
        #pragma unroll
        for (int s = 0; s < 4; ++s) {
            bf[s].q[0] = *(const uint4*)&sB[p][s * 16 + ncol][khi];
            bf[s].q[1] = *(const uint4*)&sB[p][s * 16 + ncol][16 + khi];
        }
        acc[0] = __builtin_amdgcn_wmma_f32_16x16x32_bf16(false, af.v, false, bf[0].v,
                                                         (short)0, acc[0], false, false);
        acc[1] = __builtin_amdgcn_wmma_f32_16x16x32_bf16(false, af.v, false, bf[1].v,
                                                         (short)0, acc[1], true, false);
        acc[2] = __builtin_amdgcn_wmma_f32_16x16x32_bf16(false, af.v, false, bf[2].v,
                                                         (short)0, acc[2], true, false);
        acc[3] = __builtin_amdgcn_wmma_f32_16x16x32_bf16(false, af.v, false, bf[3].v,
                                                         (short)0, acc[3], true, false);
        __syncthreads();                                // done reading buf p
    }
#else
    // fallback: simple cooperative staging (single buffer)
    const int ar0 = tid >> 2;
    const int ac0 = (tid & 3) * 8;
    for (int i = 0; i < nch; ++i) {
        const int kn = i * GTK;
        *(uint4*)&sA[0][ar0][ac0]      = *(const uint4*)&A[(size_t)(m0 + ar0) * K + kn + ac0];
        *(uint4*)&sA[0][ar0 + 64][ac0] = *(const uint4*)&A[(size_t)(m0 + ar0 + 64) * K + kn + ac0];
        *(uint4*)&sB[0][ar0][ac0]      = *(const uint4*)&Bt[(size_t)(n0 + ar0) * K + kn + ac0];
        __syncthreads();
        FragU af;
        af.q[0] = *(const uint4*)&sA[0][mrow][khi];
        af.q[1] = *(const uint4*)&sA[0][mrow][16 + khi];
        #pragma unroll
        for (int s = 0; s < 4; ++s) {
            FragU bf;
            bf.q[0] = *(const uint4*)&sB[0][s * 16 + ncol][khi];
            bf.q[1] = *(const uint4*)&sB[0][s * 16 + ncol][16 + khi];
            acc[s] = __builtin_amdgcn_wmma_f32_16x16x32_bf16(false, af.v, false, bf.v,
                                                             (short)0, acc[s], false, false);
        }
        __syncthreads();
    }
#endif

    const int crow = m0 + wave * 16 + (lane >> 4) * 8;
    const int cc0  = n0 + (lane & 15);
    #pragma unroll
    for (int s = 0; s < 4; ++s) {
        const int cc = cc0 + s * 16;
        const float bb = bias ? bias[cc] : 0.f;
        #pragma unroll
        for (int r = 0; r < 8; ++r) {
            float v = apply_act(acc[s][r] + bb, act);
            C[(size_t)(crow + r) * N + cc] = v;
            if (Cb) Cb[(size_t)(crow + r) * N + cc] = (__bf16)v;
        }
    }
}

// ---------------------------------------------------------------------------
// Small helpers
// ---------------------------------------------------------------------------
__global__ void k_cvt_bf16(const float* __restrict__ s, __bf16* __restrict__ d, long n)
{
    long i = (long)blockIdx.x * blockDim.x + threadIdx.x;
    if (i < n) d[i] = (__bf16)s[i];
}

__global__ void k_fill(float* __restrict__ d, float v, long n)
{
    long i = (long)blockIdx.x * blockDim.x + threadIdx.x;
    if (i < n) d[i] = v;
}

__global__ void k_bias_act(float* __restrict__ d, const float* __restrict__ bias,
                           long n, int C, int act)
{
    long i = (long)blockIdx.x * blockDim.x + threadIdx.x;
    if (i >= n) return;
    float v = d[i] + (bias ? bias[i % C] : 0.f);
    d[i] = apply_act(v, act);
}

// wave-per-row GEMM for small output widths; optional concat second operand
template<int NS>
__global__ void __launch_bounds__(32)
k_rowgemm(const float* __restrict__ A, int K,
          const float* __restrict__ A2, int K2,
          const float* __restrict__ Bt, int ldb,
          const float* __restrict__ bias,
          float* __restrict__ C)
{
    const int n = blockIdx.x;
    const int lane = threadIdx.x;
    float acc[NS];
    #pragma unroll
    for (int j = 0; j < NS; ++j) acc[j] = 0.f;
    const int KT = K + K2;
    for (int k = lane; k < KT; k += 32) {
        float a = (k < K) ? A[(size_t)n * K + k] : A2[(size_t)n * K2 + (k - K)];
        #pragma unroll
        for (int j = 0; j < NS; ++j) acc[j] += a * Bt[(size_t)j * ldb + k];
    }
    #pragma unroll
    for (int j = 0; j < NS; ++j) {
        float v = acc[j];
        for (int off = 16; off > 0; off >>= 1) v += __shfl_down(v, off);
        if (lane == 0) C[(size_t)n * NS + j] = v + (bias ? bias[j] : 0.f);
    }
}

template<int C>
__global__ void k_softmax(float* __restrict__ x, float scale, int n)
{
    int i = blockIdx.x * blockDim.x + threadIdx.x;
    if (i >= n) return;
    float* r = x + (size_t)i * C;
    float m = -1e30f;
    #pragma unroll
    for (int j = 0; j < C; ++j) m = fmaxf(m, r[j] * scale);
    float v[C], s = 0.f;
    #pragma unroll
    for (int j = 0; j < C; ++j) { v[j] = __expf(r[j] * scale - m); s += v[j]; }
    #pragma unroll
    for (int j = 0; j < C; ++j) r[j] = v[j] / s;
}

template<int C>
__global__ void k_ln_small(const float* __restrict__ in, const float* __restrict__ g,
                           const float* __restrict__ b, float* __restrict__ out, int n)
{
    int i = blockIdx.x * blockDim.x + threadIdx.x;
    if (i >= n) return;
    const float* r = in + (size_t)i * C;
    float m = 0.f;
    #pragma unroll
    for (int j = 0; j < C; ++j) m += r[j];
    m /= C;
    float v = 0.f;
    #pragma unroll
    for (int j = 0; j < C; ++j) { float d = r[j] - m; v += d * d; }
    float inv = rsqrtf(v / C + 1e-5f);
    #pragma unroll
    for (int j = 0; j < C; ++j) out[(size_t)i * C + j] = (r[j] - m) * inv * g[j] + b[j];
}

// LN over 10 expert outputs + weighted accumulate into expert_logits
__global__ void k_ln10_accum(const float* __restrict__ eo, const float* __restrict__ g,
                             const float* __restrict__ b, const float* __restrict__ sg,
                             int iexp, float* __restrict__ accum, int n)
{
    int i = blockIdx.x * blockDim.x + threadIdx.x;
    if (i >= n) return;
    const float* r = eo + (size_t)i * 10;
    float m = 0.f;
    #pragma unroll
    for (int j = 0; j < 10; ++j) m += r[j];
    m *= 0.1f;
    float v = 0.f;
    #pragma unroll
    for (int j = 0; j < 10; ++j) { float d = r[j] - m; v += d * d; }
    float inv = rsqrtf(v * 0.1f + 1e-5f);
    float w = sg[(size_t)i * 10 + iexp];
    #pragma unroll
    for (int j = 0; j < 10; ++j)
        accum[(size_t)i * 10 + j] += w * ((r[j] - m) * inv * g[j] + b[j]);
}

// LayerNorm over 1024, emits bf16 directly (consumed only by WMMA GEMM)
__global__ __launch_bounds__(256)
void k_layernorm(const float* __restrict__ in, const float* __restrict__ g,
                 const float* __restrict__ b, __bf16* __restrict__ outb, int C)
{
    __shared__ float red[256];
    const int n = blockIdx.x, tid = threadIdx.x;
    const float* r = in + (size_t)n * C;
    float s = 0.f;
    for (int k = tid; k < C; k += 256) s += r[k];
    red[tid] = s; __syncthreads();
    for (int o = 128; o > 0; o >>= 1) { if (tid < o) red[tid] += red[tid + o]; __syncthreads(); }
    float mean = red[0] / C; __syncthreads();
    float s2 = 0.f;
    for (int k = tid; k < C; k += 256) { float d = r[k] - mean; s2 += d * d; }
    red[tid] = s2; __syncthreads();
    for (int o = 128; o > 0; o >>= 1) { if (tid < o) red[tid] += red[tid + o]; __syncthreads(); }
    float inv = rsqrtf(red[0] / C + 1e-5f);
    for (int k = tid; k < C; k += 256)
        outb[(size_t)n * C + k] = (__bf16)((r[k] - mean) * inv * g[k] + b[k]);
}

// mem_ctx from factorized memory: coeff[n,24,5] (col0 = init weight, 1..4 = wv_t)
__global__ __launch_bounds__(128)
void k_mem_ctx(const float* __restrict__ attn, const float* __restrict__ coeff,
               const float* __restrict__ init, const float* __restrict__ wvbuf,
               float* __restrict__ ctx, int Nrows, int C)
{
    __shared__ float w0[24];
    __shared__ float st[4];
    const int n = blockIdx.x, tid = threadIdx.x;
    if (tid < 24) w0[tid] = attn[(size_t)n * 24 + tid] * coeff[((size_t)n * 24 + tid) * 5];
    if (tid >= 24 && tid < 28) {
        int t = tid - 24;
        float s = 0.f;
        for (int m = 0; m < 24; ++m)
            s += attn[(size_t)n * 24 + m] * coeff[((size_t)n * 24 + m) * 5 + 1 + t];
        st[t] = s;
    }
    __syncthreads();
    for (int d = tid; d < C; d += 128) {
        float v = 0.f;
        #pragma unroll
        for (int m = 0; m < 24; ++m) v += w0[m] * init[(size_t)m * C + d];
        #pragma unroll
        for (int t = 0; t < 4; ++t) v += st[t] * wvbuf[((size_t)t * Nrows + n) * C + d];
        ctx[(size_t)n * C + d] = v;
    }
}

// depth attention over cached k/v bank (s entries, s <= 3)
__global__ __launch_bounds__(128)
void k_depth_attn(const float* __restrict__ q, const float* __restrict__ kc,
                  const float* __restrict__ vc, int s, float scale,
                  float* __restrict__ ctx, int Nrows, int C)
{
    __shared__ float red[128];
    __shared__ float att[4];
    const int n = blockIdx.x, tid = threadIdx.x;
    for (int j = 0; j < s; ++j) {
        float p = 0.f;
        for (int d = tid; d < C; d += 128)
            p += q[(size_t)n * C + d] * kc[((size_t)j * Nrows + n) * C + d];
        red[tid] = p; __syncthreads();
        for (int o = 64; o > 0; o >>= 1) { if (tid < o) red[tid] += red[tid + o]; __syncthreads(); }
        if (tid == 0) att[j] = red[0] * scale;
        __syncthreads();
    }
    if (tid == 0) {
        float m = -1e30f;
        for (int j = 0; j < s; ++j) m = fmaxf(m, att[j]);
        float ssum = 0.f;
        for (int j = 0; j < s; ++j) { att[j] = __expf(att[j] - m); ssum += att[j]; }
        for (int j = 0; j < s; ++j) att[j] /= ssum;
    }
    __syncthreads();
    for (int d = tid; d < C; d += 128) {
        float v = 0.f;
        for (int j = 0; j < s; ++j) v += att[j] * vc[((size_t)j * Nrows + n) * C + d];
        ctx[(size_t)n * C + d] = v;
    }
}

// gate: mix-softmax, expert-softmax, budget argmax, top-4 sparse renorm, surprise
__global__ void k_gate(const float* __restrict__ sub, const float* __restrict__ mixlog,
                       const float* __restrict__ ebias, const float* __restrict__ budlog,
                       float* __restrict__ sg, float* __restrict__ surprise, int n)
{
    int i = blockIdx.x * blockDim.x + threadIdx.x;
    if (i >= n) return;
    const float* sb = sub + (size_t)i * 40;
    const float* ml = mixlog + (size_t)i * 4;
    float mm = -1e30f;
    #pragma unroll
    for (int j = 0; j < 4; ++j) mm = fmaxf(mm, ml[j]);
    float mw[4], msum = 0.f;
    #pragma unroll
    for (int j = 0; j < 4; ++j) { mw[j] = __expf(ml[j] - mm); msum += mw[j]; }
    #pragma unroll
    for (int j = 0; j < 4; ++j) mw[j] /= msum;

    float gl[10];
    #pragma unroll
    for (int e = 0; e < 10; ++e) {
        float v = ebias[e];
        #pragma unroll
        for (int s = 0; s < 4; ++s) v += mw[s] * sb[s * 10 + e];
        gl[e] = v;
    }
    float gm = -1e30f;
    #pragma unroll
    for (int e = 0; e < 10; ++e) gm = fmaxf(gm, gl[e]);
    float p[10], psum = 0.f;
    #pragma unroll
    for (int e = 0; e < 10; ++e) { p[e] = __expf(gl[e] - gm); psum += p[e]; }
    #pragma unroll
    for (int e = 0; e < 10; ++e) p[e] /= psum;

    const float* bl = budlog + (size_t)i * 4;
    int bud = 0; float bm = bl[0];
    #pragma unroll
    for (int j = 1; j < 4; ++j) if (bl[j] > bm) { bm = bl[j]; bud = j; }
    bud += 1;

    float p2[10];
    #pragma unroll
    for (int e = 0; e < 10; ++e) p2[e] = p[e];
    float vals[4]; int idx[4];
    #pragma unroll
    for (int t = 0; t < 4; ++t) {
        int bi = 0; float bv = p2[0];
        #pragma unroll
        for (int e = 1; e < 10; ++e) if (p2[e] > bv) { bv = p2[e]; bi = e; }
        vals[t] = bv; idx[t] = bi; p2[bi] = -1.f;
    }
    float outp[10];
    #pragma unroll
    for (int e = 0; e < 10; ++e) outp[e] = 0.f;
    float ssum = 0.f;
    #pragma unroll
    for (int t = 0; t < 4; ++t)
        if (t < bud) { outp[idx[t]] = vals[t]; ssum += vals[t]; }
    ssum = fmaxf(ssum, 1e-6f);
    float mx = 0.f;
    #pragma unroll
    for (int e = 0; e < 10; ++e) {
        float v = outp[e] / ssum;
        sg[(size_t)i * 10 + e] = v;
        mx = fmaxf(mx, v);
    }
    surprise[i] = 1.f - mx;
}

__global__ void k_coeff_init(float* __restrict__ c, int nm)
{
    int i = blockIdx.x * blockDim.x + threadIdx.x;
    if (i >= nm) return;
    float* p = c + (size_t)i * 5;
    p[0] = 1.f; p[1] = 0.f; p[2] = 0.f; p[3] = 0.f; p[4] = 0.f;
}

__global__ void k_coeff_update(float* __restrict__ c, const float* __restrict__ wg,
                               const float* __restrict__ sur, int step, int nm, int MM)
{
    int i = blockIdx.x * blockDim.x + threadIdx.x;
    if (i >= nm) return;
    int row = i / MM;
    float g = sur[row] * wg[i];
    float* p = c + (size_t)i * 5;
    #pragma unroll
    for (int j = 0; j < 5; ++j) p[j] *= (1.f - g);
    p[1 + step] = g;
}

__global__ void k_enrich(const float* __restrict__ c, const float* __restrict__ m,
                         const float* __restrict__ d, const float* __restrict__ s,
                         float* __restrict__ o, long n)
{
    long i = (long)blockIdx.x * blockDim.x + threadIdx.x;
    if (i < n) o[i] = c[i] + 0.34f * m[i] + 0.22f * d[i] + 0.18f * s[i];
}

__global__ void k_lincomb3(const float* __restrict__ c, const float* __restrict__ m,
                           const float* __restrict__ d, float* __restrict__ o, long n)
{
    long i = (long)blockIdx.x * blockDim.x + threadIdx.x;
    if (i < n) o[i] = c[i] + 0.2f * m[i] + 0.1f * d[i];
}

// residual add; emits fp32 current AND its bf16 copy (feeds next WMMA GEMMs)
__global__ void k_resid(const float* __restrict__ enr, const float* __restrict__ f2,
                        const float* __restrict__ bias, float* __restrict__ cur,
                        __bf16* __restrict__ curb, long n, int C)
{
    long i = (long)blockIdx.x * blockDim.x + threadIdx.x;
    if (i >= n) return;
    float v = enr[i] + f2[i] + bias[i % C];
    cur[i] = v;
    curb[i] = (__bf16)v;
}

__global__ void k_ssmb(const float* __restrict__ t, const float* __restrict__ Bt,
                       const float* __restrict__ bias, float* __restrict__ out, int n)
{
    int idx = blockIdx.x * blockDim.x + threadIdx.x;
    if (idx >= n * 1024) return;
    int i = idx & 1023;
    int r = idx >> 10;
    float s = bias[i];
    #pragma unroll
    for (int k = 0; k < 40; ++k) s += t[(size_t)r * 40 + k] * Bt[(size_t)i * 40 + k];
    out[idx] = s;
}

__global__ void k_step_accum(const float* __restrict__ el, const float* __restrict__ m10,
                             const float* __restrict__ d10, const float* __restrict__ hl,
                             float* __restrict__ cum, float* __restrict__ total, int n)
{
    int i = blockIdx.x * blockDim.x + threadIdx.x;
    if (i >= n) return;
    float rem = 1.f - cum[i];
    float h = 1.f / (1.f + __expf(-hl[i]));
    #pragma unroll
    for (int j = 0; j < 10; ++j) {
        size_t o = (size_t)i * 10 + j;
        total[o] += rem * (el[o] + 0.22f * m10[o] + 0.14f * d10[o]);
    }
    cum[i] += rem * h;
}

__global__ void k_final(const float* __restrict__ base, const float* __restrict__ sh,
                        const float* __restrict__ total, const float* __restrict__ last10,
                        const float* __restrict__ ss, const float* __restrict__ alpha,
                        const float* __restrict__ beta, float* __restrict__ out, long n)
{
    long i = (long)blockIdx.x * blockDim.x + threadIdx.x;
    if (i >= n) return;
    out[i] = base[i] + ss[0] * sh[i] + (alpha[0] + 1e-4f) * (total[i] * 0.25f)
           + beta[0] * last10[i];
}

// ---------------------------------------------------------------------------
// Host orchestration
// ---------------------------------------------------------------------------
static inline int g256(long n) { return (int)((n + 255) / 256); }

extern "C" void kernel_launch(void* const* d_in, const int* in_sizes, int n_in,
                              void* d_out, int out_size, void* d_ws, size_t ws_size,
                              hipStream_t stream)
{
    (void)in_sizes; (void)n_in; (void)out_size; (void)ws_size;

    const int N = 2048, IN = 1024, OUT = 10, E = 10, MM = 24, S = 4,
              DST = 40, INNER = 768, SH = 2048, DMAX = 3328;
    static const int DIMS[10] = {1536, 2048, 2560, 3072, 1792, 2304, 2816, 2048, 2560, 3328};
    const float msc = 0.03125f; // 1024^-0.5

    const float* x            = (const float*)d_in[0];
    const float* weight       = (const float*)d_in[1];
    const float* bias         = (const float*)d_in[2];
    const float* shared_up_w  = (const float*)d_in[3];
    const float* shared_down_w= (const float*)d_in[4];
    const float* shared_ng    = (const float*)d_in[5];
    const float* shared_nb    = (const float*)d_in[6];
    const float* shared_scale = (const float*)d_in[7];
    const float* memory_keys  = (const float*)d_in[8];
    const float* mem_init     = (const float*)d_in[9];
    const float* mem_q_w      = (const float*)d_in[10];
    const float* mem_out_w    = (const float*)d_in[11];
    const float* wgate_w      = (const float*)d_in[12];
    const float* wgate_b      = (const float*)d_in[13];
    const float* wval_w       = (const float*)d_in[14];
    const float* wval_b       = (const float*)d_in[15];
    const float* dq_w         = (const float*)d_in[16];
    const float* dk_w         = (const float*)d_in[17];
    const float* dv_w         = (const float*)d_in[18];
    const float* dout_w       = (const float*)d_in[19];
    const float* cln_g        = (const float*)d_in[20];
    const float* cln_b        = (const float*)d_in[21];
    const float* fc1_w        = (const float*)d_in[22];
    const float* fc1_b        = (const float*)d_in[23];
    const float* fc2_w        = (const float*)d_in[24];
    const float* fc2_b        = (const float*)d_in[25];
    const float* ssm_a_w      = (const float*)d_in[26];
    const float* ssm_a_b      = (const float*)d_in[27];
    const float* ssm_b_w      = (const float*)d_in[28];
    const float* ssm_b_b      = (const float*)d_in[29];
    const float* mor_sub_w    = (const float*)d_in[30];
    const float* mor_sub_b    = (const float*)d_in[31];
    const float* mor_mix_w    = (const float*)d_in[32];
    const float* mor_mix_b    = (const float*)d_in[33];
    const float* expert_bias  = (const float*)d_in[34];
    const float* budget_w     = (const float*)d_in[35];
    const float* budget_b     = (const float*)d_in[36];
    const float* experts_up   = (const float*)d_in[37];
    const float* experts_down = (const float*)d_in[38];
    const float* en_g         = (const float*)d_in[39];
    const float* en_b         = (const float*)d_in[40];
    const float* halt_w       = (const float*)d_in[41];
    const float* halt_b       = (const float*)d_in[42];
    const float* alpha        = (const float*)d_in[43];
    const float* beta         = (const float*)d_in[44];
    float* out = (float*)d_out;

    // ---- workspace carve -------------------------------------------------
    char* wsp = (char*)d_ws;
    size_t off = 0;
    auto ALLOC = [&](size_t bytes) -> char* {
        char* p = wsp + off;
        off = (off + bytes + 255) & ~(size_t)255;
        return p;
    };
    auto F = [&](long n) -> float*  { return (float*)ALLOC((size_t)n * 4); };
    auto B = [&](long n) -> __bf16* { return (__bf16*)ALLOC((size_t)n * 2); };

    __bf16* wb_shup = B((long)SH * IN);
    __bf16* wb_mqw  = B((long)IN * IN);
    __bf16* wb_dqw  = B((long)IN * IN);
    __bf16* wb_dkw  = B((long)IN * IN);
    __bf16* wb_dvw  = B((long)IN * IN);
    __bf16* wb_wvw  = B((long)IN * IN);
    __bf16* wb_fc1  = B((long)S * INNER * IN);
    __bf16* wb_fc2  = B((long)S * IN * INNER);
    __bf16* wb_eu   = B((long)E * DMAX * IN);
    __bf16* curb    = B((long)N * IN);
    __bf16* lnb     = B((long)N * IN);
    __bf16* hb      = B((long)N * INNER);

    float* cur      = F((long)N * IN);
    float* mq       = F((long)N * IN);
    float* qv       = F((long)N * IN);
    float* mem_ctx  = F((long)N * IN);
    float* depth_ctx= F((long)N * IN);
    float* ssm_corr = F((long)N * IN);
    float* enriched = F((long)N * IN);
    float* fc1out   = F((long)N * INNER);
    float* fc2out   = F((long)N * IN);
    float* tmpv     = F((long)N * IN);
    float* shared_h = F((long)N * SH);
    float* kcache   = F(4L * N * IN);
    float* vcache   = F(4L * N * IN);
    float* wvbuf    = F(4L * N * IN);
    float* upout    = F((long)N * DMAX);
    float* basev    = F((long)N * OUT);
    float* shared10 = F((long)N * OUT);
    float* sharedv  = F((long)N * OUT);
    float* eo10     = F((long)N * OUT);
    float* mem10    = F((long)N * OUT);
    float* dep10    = F((long)N * OUT);
    float* elog     = F((long)N * OUT);
    float* total    = F((long)N * OUT);
    float* attn     = F((long)N * MM);
    float* wg       = F((long)N * MM);
    float* ssm_t    = F((long)N * DST);
    float* subb     = F((long)N * 40);
    float* mixlog   = F((long)N * 4);
    float* budlog   = F((long)N * 4);
    float* sgbuf    = F((long)N * E);
    float* sur      = F((long)N);
    float* hl       = F((long)N);
    float* cumh     = F((long)N);
    float* coeff    = F((long)N * MM * 5);

    auto cvt = [&](const float* src, __bf16* dst, long n) {
        k_cvt_bf16<<<g256(n), 256, 0, stream>>>(src, dst, n);
    };
    auto gemm = [&](const __bf16* A, const __bf16* Bt, float* C, __bf16* Cb,
                    const float* bi, int act, int Nc, int K) {
        dim3 grid(Nc / GTN, N / GTM);
        k_wmma_gemm_bf16<<<grid, 256, 0, stream>>>(A, Bt, C, Cb, bi, act, N, Nc, K);
    };
    auto gemm_tdm = [&](const __bf16* A, const __bf16* Bt, float* C, int act,
                        int Nc, int K) {
        dim3 grid(Nc / GTN, N / GTM);
        k_wmma_gemm_tdm<<<grid, 256, 0, stream>>>(A, Bt, C, nullptr, nullptr, act, N, Nc, K);
    };

    // ---- weight conversion to bf16 (per launch; deterministic) -----------
    cvt(shared_up_w, wb_shup, (long)SH * IN);
    cvt(mem_q_w, wb_mqw, (long)IN * IN);
    cvt(dq_w, wb_dqw, (long)IN * IN);
    cvt(dk_w, wb_dkw, (long)IN * IN);
    cvt(dv_w, wb_dvw, (long)IN * IN);
    cvt(wval_w, wb_wvw, (long)IN * IN);
    cvt(fc1_w, wb_fc1, (long)S * INNER * IN);
    cvt(fc2_w, wb_fc2, (long)S * IN * INNER);
    cvt(experts_up, wb_eu, (long)E * DMAX * IN);

    // ---- init state -------------------------------------------------------
    k_fill<<<g256((long)N * OUT), 256, 0, stream>>>(total, 0.f, (long)N * OUT);
    k_fill<<<g256(N), 256, 0, stream>>>(cumh, 0.f, N);
    k_fill<<<g256(4L * N * IN), 256, 0, stream>>>(wvbuf, 0.f, 4L * N * IN);
    k_coeff_init<<<g256((long)N * MM), 256, 0, stream>>>(coeff, N * MM);
    hipMemcpyAsync(cur, x, (size_t)N * IN * 4, hipMemcpyDeviceToDevice, stream);
    cvt(x, curb, (long)N * IN);          // curb tracks cur from here on

    // ---- base + shared path ----------------------------------------------
    k_rowgemm<10><<<N, 32, 0, stream>>>(x, IN, x, 0, weight, IN, bias, basev);
    gemm(curb, wb_shup, shared_h, nullptr, nullptr, 0 /*silu fused*/, SH, IN);
    k_rowgemm<10><<<N, 32, 0, stream>>>(shared_h, SH, shared_h, 0, shared_down_w, SH,
                                        nullptr, shared10);
    k_ln_small<10><<<g256(N), 256, 0, stream>>>(shared10, shared_ng, shared_nb, sharedv, N);

    // ---- recursive step loop ----------------------------------------------
    for (int t = 0; t < S; ++t) {
        // memory attention (factorized mem_vals)
        gemm(curb, wb_mqw, mq, nullptr, nullptr, 8, IN, IN);
        k_rowgemm<24><<<N, 32, 0, stream>>>(mq, IN, mq, 0, memory_keys, IN, nullptr, attn);
        k_softmax<24><<<g256(N), 256, 0, stream>>>(attn, msc, N);
        k_mem_ctx<<<N, 128, 0, stream>>>(attn, coeff, mem_init, wvbuf, mem_ctx, N, IN);

        // depth attention over cached bank
        if (t == 0) {
            k_fill<<<g256((long)N * IN), 256, 0, stream>>>(depth_ctx, 0.f, (long)N * IN);
        } else {
            gemm(curb, wb_dqw, qv, nullptr, nullptr, 8, IN, IN);
            k_depth_attn<<<N, 128, 0, stream>>>(qv, kcache, vcache, t, msc, depth_ctx, N, IN);
        }

        // SSM correction
        k_rowgemm<40><<<N, 32, 0, stream>>>(cur, IN, cur, 0, ssm_a_w, IN, ssm_a_b, ssm_t);
        k_bias_act<<<g256((long)N * DST), 256, 0, stream>>>(ssm_t, nullptr, (long)N * DST, 1, 5);
        k_ssmb<<<g256((long)N * IN), 256, 0, stream>>>(ssm_t, ssm_b_w, ssm_b_b, ssm_corr, N);

        // cell MLP (LN emits bf16; fc1 fuses bias+gelu+bf16; resid emits bf16)
        k_enrich<<<g256((long)N * IN), 256, 0, stream>>>(cur, mem_ctx, depth_ctx, ssm_corr,
                                                         enriched, (long)N * IN);
        k_layernorm<<<N, 256, 0, stream>>>(enriched, cln_g + (size_t)t * IN,
                                           cln_b + (size_t)t * IN, lnb, IN);
        gemm(lnb, wb_fc1 + (size_t)t * INNER * IN, fc1out, hb,
             fc1_b + (size_t)t * INNER, 1 /*gelu*/, INNER, IN);
        gemm(hb, wb_fc2 + (size_t)t * IN * INNER, fc2out, nullptr, nullptr, 8, IN, INNER);
        k_resid<<<g256((long)N * IN), 256, 0, stream>>>(enriched, fc2out,
                                                        fc2_b + (size_t)t * IN, cur, curb,
                                                        (long)N * IN, IN);

        // append to bank: cache k/v projections of new current
        gemm(curb, wb_dkw, kcache + (size_t)t * N * IN, nullptr, nullptr, 8, IN, IN);
        gemm(curb, wb_dvw, vcache + (size_t)t * N * IN, nullptr, nullptr, 8, IN, IN);

        // routing
        k_rowgemm<40><<<N, 32, 0, stream>>>(cur, IN, cur, 0, mor_sub_w, IN, mor_sub_b, subb);
        k_rowgemm<4><<<N, 32, 0, stream>>>(cur, IN, cur, 0, mor_mix_w, IN, mor_mix_b, mixlog);
        k_rowgemm<4><<<N, 32, 0, stream>>>(cur, IN, cur, 0, budget_w, IN, budget_b, budlog);
        k_gate<<<g256(N), 256, 0, stream>>>(subb, mixlog, expert_bias, budlog, sgbuf, sur, N);

        // experts: TDM-fed WMMA up-proj (act fused) -> wave-reduce down-proj
        k_fill<<<g256((long)N * OUT), 256, 0, stream>>>(elog, 0.f, (long)N * OUT);
        for (int i = 0; i < E; ++i) {
            int d = DIMS[i];
            gemm_tdm(curb, wb_eu + (size_t)i * DMAX * IN, upout, i % 8, d, IN);
            k_rowgemm<10><<<N, 32, 0, stream>>>(upout, d, upout, 0,
                                                experts_down + (size_t)i * OUT * DMAX, DMAX,
                                                nullptr, eo10);
            k_ln10_accum<<<g256(N), 256, 0, stream>>>(eo10, en_g + (size_t)i * OUT,
                                                      en_b + (size_t)i * OUT, sgbuf, i, elog, N);
        }

        // step logits side terms + halting
        k_rowgemm<10><<<N, 32, 0, stream>>>(mem_ctx, IN, mem_ctx, 0, mem_out_w, IN, nullptr,
                                            mem10);
        k_rowgemm<10><<<N, 32, 0, stream>>>(depth_ctx, IN, depth_ctx, 0, dout_w, IN, nullptr,
                                            dep10);
        k_lincomb3<<<g256((long)N * IN), 256, 0, stream>>>(cur, mem_ctx, depth_ctx, tmpv,
                                                           (long)N * IN);
        k_rowgemm<1><<<N, 32, 0, stream>>>(tmpv, IN, tmpv, 0, halt_w + (size_t)t * IN, IN,
                                           halt_b + t, hl);

        // memory write (factorized): wv_t (bias fused), write gates, coeff update
        gemm(curb, wb_wvw, wvbuf + (size_t)t * N * IN, nullptr, wval_b, 8, IN, IN);
        k_rowgemm<24><<<N, 32, 0, stream>>>(cur, IN, mem_ctx, IN, wgate_w, 2 * IN, wgate_b, wg);
        k_softmax<24><<<g256(N), 256, 0, stream>>>(wg, 1.f, N);
        k_coeff_update<<<g256((long)N * MM), 256, 0, stream>>>(coeff, wg, sur, t, N * MM, MM);

        // accumulate halted output
        k_step_accum<<<g256(N), 256, 0, stream>>>(elog, mem10, dep10, hl, cumh, total, N);
    }

    // ---- final combine -----------------------------------------------------
    k_rowgemm<10><<<N, 32, 0, stream>>>(cur, IN, cur, 0, mem_out_w, IN, nullptr, eo10);
    k_final<<<g256((long)N * OUT), 256, 0, stream>>>(basev, sharedv, total, eo10,
                                                     shared_scale, alpha, beta, out,
                                                     (long)N * OUT);
}
